// KNNGNN_39333310496825
// MI455X (gfx1250) — matrix operand order repaired and checked by reading, not compiled, not run on hardware
//
#include <hip/hip_runtime.h>

typedef __attribute__((ext_vector_type(16))) _Float16 v16h;
typedef __attribute__((ext_vector_type(8)))  float    v8f;

#define NN      10000
#define FF      512
#define HHEADS  8
#define CC      128
#define PROJ    64
#define NCLS    40
#define KKNN    35
#define EE      160000
#define NEG_SLOPE 0.2f

// ---------------------------------------------------------------------------
// WMMA GEMM:  C[M,Nc] = act( A[M,K](f16,row-major) @ B (f16, given TRANSPOSED:
// Bt[NcPad][K]) + bias ).  block = 128 threads (4 waves); each wave owns one
// 16x16 tile; block covers 16 rows x 64 cols.  Requires M%16==0, K%64==0,
// gridDim.x*64 <= NcPad (Bt column-padded with zeros).
// Both A and Bt fragments are contiguous 32B runs per lane -> b128 loads.
// ---------------------------------------------------------------------------
__global__ __launch_bounds__(128)
void kg_gemm_wmma(const _Float16* __restrict__ A, const _Float16* __restrict__ Bt,
                  const float* __restrict__ bias, float* __restrict__ C,
                  int M, int K, int Nc, int relu_flag)
{
    const int lane  = threadIdx.x & 31;
    const int wave  = threadIdx.x >> 5;
    const int m     = lane & 15;
    const int khalf = lane >> 4;
    const int tileM = blockIdx.y << 4;
    const int tileN = (blockIdx.x << 6) + (wave << 4);
    const int row   = tileM + m;          // A row for this lane
    const int col   = tileN + m;          // B/D column for this lane (< NcPad)

    const _Float16* Ap = A  + (size_t)row * K + khalf * 16;  // a[e]=Ap[k0+e]
    const _Float16* Bp = Bt + (size_t)col * K + khalf * 16;  // b[e]=Bp[k0+e]

    v8f acc = {};
    for (int k0 = 0; k0 < K; k0 += 64) {
        v16h a0, b0, a1, b1;
        #pragma unroll
        for (int e = 0; e < 8; ++e) { a0[e] = Ap[k0 + e];      a0[8 + e] = Ap[k0 + 8 + e]; }
        #pragma unroll
        for (int e = 0; e < 16; ++e) b0[e] = Bp[k0 + e];
        #pragma unroll
        for (int e = 0; e < 8; ++e) { a1[e] = Ap[k0 + 32 + e]; a1[8 + e] = Ap[k0 + 40 + e]; }
        #pragma unroll
        for (int e = 0; e < 16; ++e) b1[e] = Bp[k0 + 32 + e];

        acc = __builtin_amdgcn_wmma_f32_16x16x32_f16(
                  false, a0, false, b0, (short)0, acc, false, false);
        acc = __builtin_amdgcn_wmma_f32_16x16x32_f16(
                  false, a1, false, b1, (short)0, acc, false, false);
    }

    if (col < Nc) {
        const float bv = bias ? bias[col] : 0.0f;
        #pragma unroll
        for (int v = 0; v < 8; ++v) {
            const int r = tileM + khalf * 8 + v;
            float val = acc[v] + bv;
            if (relu_flag) val = fmaxf(val, 0.0f);
            C[(size_t)r * Nc + col] = val;
        }
    }
}

// ---------------------------------------------------------------------------
__global__ void kg_cvt_h(const float* __restrict__ in, _Float16* __restrict__ out, size_t n)
{
    size_t t = (size_t)blockIdx.x * blockDim.x + threadIdx.x;
    if (t < n) out[t] = (_Float16)in[t];
}

// weight [K][Nc] f32  ->  wt [NcPad][K] f16 (transposed, zero-padded columns)
__global__ void kg_cvt_tr(const float* __restrict__ w, _Float16* __restrict__ wt,
                          int K, int Nc, int NcPad)
{
    int t = blockIdx.x * blockDim.x + threadIdx.x;
    if (t >= NcPad * K) return;
    const int n = t / K, k = t - n * K;
    wt[t] = (n < Nc) ? (_Float16)w[(size_t)k * Nc + n] : (_Float16)0;
}

__global__ void kg_fill(float* __restrict__ p, float v, size_t n)
{
    size_t t = (size_t)blockIdx.x * blockDim.x + threadIdx.x;
    if (t < n) p[t] = v;
}

// per-(node,head) attention logits: als = <xh[n,h,:], a_src[h,:]>, ald likewise
__global__ void kg_attn_logits(const float* __restrict__ xh,
                               const float* __restrict__ a_s, const float* __restrict__ a_d,
                               float* __restrict__ als, float* __restrict__ ald,
                               int n_nodes, int heads)
{
    int t = blockIdx.x * blockDim.x + threadIdx.x;
    if (t >= n_nodes * heads) return;
    const int h = t % heads;
    const float* xr  = xh  + (size_t)t * CC;   // [n, heads, C] flattened
    const float* asv = a_s + h * CC;
    const float* adv = a_d + h * CC;
    float s = 0.f, d = 0.f;
    for (int c = 0; c < CC; ++c) { const float xv = xr[c]; s += xv * asv[c]; d += xv * adv[c]; }
    als[t] = s; ald[t] = d;
}

__device__ inline void kg_atomicMaxF(float* addr, float val)
{
    unsigned int* a = (unsigned int*)addr;
    unsigned int old = *a;
    while (true) {
        if (__uint_as_float(old) >= val) return;
        unsigned int prev = atomicCAS(a, old, __float_as_uint(val));
        if (prev == old) return;
        old = prev;
    }
}

__global__ void kg_edge_max(const int* __restrict__ src, const int* __restrict__ dst,
                            const float* __restrict__ als, const float* __restrict__ ald,
                            float* __restrict__ ebuf, float* __restrict__ mbuf, int heads)
{
    int t = blockIdx.x * blockDim.x + threadIdx.x;
    if (t >= EE * heads) return;
    const int e = t / heads, h = t - e * heads;
    const int s = src[e], d = dst[e];
    float v = als[s * heads + h] + ald[d * heads + h];
    v = (v > 0.f) ? v : NEG_SLOPE * v;          // leaky_relu
    ebuf[t] = v;
    kg_atomicMaxF(mbuf + d * heads + h, v);
}

__global__ void kg_edge_exp(const int* __restrict__ dst,
                            float* __restrict__ ebuf, const float* __restrict__ mbuf,
                            float* __restrict__ dbuf, int heads)
{
    int t = blockIdx.x * blockDim.x + threadIdx.x;
    if (t >= EE * heads) return;
    const int e = t / heads, h = t - e * heads;
    const int d = dst[e];
    const float v = expf(ebuf[t] - mbuf[d * heads + h]);
    ebuf[t] = v;
    atomicAdd(dbuf + d * heads + h, v);
}

// out[dst,h,:] += xh[src,h,:] * alpha ; one thread handles 4 channels
__global__ void kg_edge_aggr(const int* __restrict__ src, const int* __restrict__ dst,
                             const float* __restrict__ ebuf, const float* __restrict__ denom,
                             const float* __restrict__ xh, float* __restrict__ out, int heads)
{
    const int groups = CC / 4;
    long long t = (long long)blockIdx.x * blockDim.x + threadIdx.x;
    const long long total = (long long)EE * heads * groups;
    if (t >= total) return;
    const int g  = (int)(t % groups);
    const long long eh = t / groups;
    const int h = (int)(eh % heads);
    const int e = (int)(eh / heads);
    const int s = src[e], d = dst[e];
    const float alpha = ebuf[eh] / denom[d * heads + h];
    const float4 xv = *(const float4*)(xh + ((size_t)s * heads + h) * CC + g * 4);
    float* o = out + ((size_t)d * heads + h) * CC + g * 4;
    atomicAdd(o + 0, xv.x * alpha);
    atomicAdd(o + 1, xv.y * alpha);
    atomicAdd(o + 2, xv.z * alpha);
    atomicAdd(o + 3, xv.w * alpha);
}

__global__ void kg_bias_relu_h(const float* __restrict__ in, const float* __restrict__ bias,
                               _Float16* __restrict__ out, size_t total, int ld)
{
    size_t t = (size_t)blockIdx.x * blockDim.x + threadIdx.x;
    if (t >= total) return;
    const float v = in[t] + bias[t % ld];
    out[t] = (_Float16)fmaxf(v, 0.0f);
}

// final = concat(x1, out2 + b2)  -> f32 + f16 copies
__global__ void kg_build_final(const float* __restrict__ x1, const float* __restrict__ out2,
                               const float* __restrict__ b2,
                               float* __restrict__ ff, _Float16* __restrict__ fh)
{
    int t = blockIdx.x * blockDim.x + threadIdx.x;
    if (t >= NN * 2 * CC) return;
    const int c = t % (2 * CC);
    const int n = t / (2 * CC);
    const float v = (c < CC) ? x1[(size_t)n * CC + c]
                             : (out2[(size_t)n * CC + (c - CC)] + b2[c - CC]);
    ff[t] = v; fh[t] = (_Float16)v;
}

__global__ void kg_sqnorm(const float* __restrict__ g, float* __restrict__ sq)
{
    int n = blockIdx.x * blockDim.x + threadIdx.x;
    if (n >= NN) return;
    const float* gr = g + (size_t)n * PROJ;
    float s = 0.f;
    for (int p = 0; p < PROJ; ++p) s += gr[p] * gr[p];
    sq[n] = s;
}

// Exact KNN (K=35, lowest-index tiebreak like lax.top_k): one block per row.
__global__ __launch_bounds__(256)
void kg_knn(const float* __restrict__ g, const float* __restrict__ sq, int* __restrict__ idx)
{
    __shared__ float d2[NN];
    __shared__ float gi[PROJ];
    __shared__ float sval[256];
    __shared__ int   sidx[256];
    const int i = blockIdx.x, tid = threadIdx.x;
    if (tid < PROJ) gi[tid] = g[(size_t)i * PROJ + tid];
    __syncthreads();
    const float sqi = sq[i];
    for (int j = tid; j < NN; j += 256) {
        const float* gj = g + (size_t)j * PROJ;
        float dot = 0.f;
        #pragma unroll
        for (int p = 0; p < PROJ; p += 4)
            dot += gi[p] * gj[p] + gi[p+1] * gj[p+1] + gi[p+2] * gj[p+2] + gi[p+3] * gj[p+3];
        d2[j] = sqi + sq[j] - 2.f * dot;
    }
    __syncthreads();
    for (int k = 0; k < KKNN; ++k) {
        float best = __builtin_inff(); int bi = NN;
        for (int j = tid; j < NN; j += 256) {
            const float v = d2[j];
            if (v < best || (v == best && j < bi)) { best = v; bi = j; }
        }
        sval[tid] = best; sidx[tid] = bi;
        __syncthreads();
        for (int s = 128; s > 0; s >>= 1) {
            if (tid < s) {
                const float ov = sval[tid + s]; const int oi = sidx[tid + s];
                if (ov < sval[tid] || (ov == sval[tid] && oi < sidx[tid])) {
                    sval[tid] = ov; sidx[tid] = oi;
                }
            }
            __syncthreads();
        }
        if (tid == 0) { const int w = sidx[0]; idx[(size_t)i * KKNN + k] = w; d2[w] = __builtin_inff(); }
        __syncthreads();
    }
}

__global__ void kg_sim(const float* __restrict__ ff, const int* __restrict__ idx,
                       float* __restrict__ sim)
{
    int t = blockIdx.x * blockDim.x + threadIdx.x;
    if (t >= NN * 2 * CC) return;
    const int f = t % (2 * CC);
    const int n = t / (2 * CC);
    const int* row = idx + (size_t)n * KKNN;
    float s = 0.f;
    for (int k = 0; k < KKNN; ++k) s += ff[(size_t)row[k] * (2 * CC) + f];
    sim[t] = s * (1.0f / KKNN);
}

__global__ void kg_build_c3(const float* __restrict__ x1, const float* __restrict__ sim,
                            _Float16* __restrict__ c3)
{
    int t = blockIdx.x * blockDim.x + threadIdx.x;
    if (t >= NN * 3 * CC) return;
    const int f = t % (3 * CC);
    const int n = t / (3 * CC);
    const float v = (f < CC) ? x1[(size_t)n * CC + f] : sim[(size_t)n * (2 * CC) + (f - CC)];
    c3[t] = (_Float16)v;
}

__global__ void kg_logsoftmax(const float* __restrict__ logits, float* __restrict__ out)
{
    int n = blockIdx.x * blockDim.x + threadIdx.x;
    if (n >= NN) return;
    const float* lr = logits + (size_t)n * NCLS;
    float mx = -__builtin_inff();
    for (int c = 0; c < NCLS; ++c) mx = fmaxf(mx, lr[c]);
    float s = 0.f;
    for (int c = 0; c < NCLS; ++c) s += expf(lr[c] - mx);
    const float ls = logf(s);
    float* orow = out + (size_t)n * NCLS;
    for (int c = 0; c < NCLS; ++c) orow[c] = lr[c] - mx - ls;
}

// ---------------------------------------------------------------------------
static inline unsigned kg_nb(long long n, int tpb) { return (unsigned)((n + tpb - 1) / tpb); }

extern "C" void kernel_launch(void* const* d_in, const int* in_sizes, int n_in,
                              void* d_out, int out_size, void* d_ws, size_t ws_size,
                              hipStream_t stream)
{
    (void)in_sizes; (void)n_in; (void)out_size; (void)ws_size;
    const float* x      = (const float*)d_in[0];
    const int*   ei     = (const int*)d_in[1];
    const float* lin1_w = (const float*)d_in[2];
    const float* lin1_b = (const float*)d_in[3];
    const float* W1     = (const float*)d_in[4];
    const float* as1    = (const float*)d_in[5];
    const float* ad1    = (const float*)d_in[6];
    const float* b1     = (const float*)d_in[7];
    const float* W2     = (const float*)d_in[8];
    const float* as2    = (const float*)d_in[9];
    const float* ad2    = (const float*)d_in[10];
    const float* b2     = (const float*)d_in[11];
    const float* proj_w = (const float*)d_in[12];
    const float* proj_b = (const float*)d_in[13];
    const float* lin_w  = (const float*)d_in[14];
    const float* lin_b  = (const float*)d_in[15];
    const int* src = ei;
    const int* dst = ei + EE;

    char* ws = (char*)d_ws;
    size_t off = 0;
    auto alloc = [&](size_t bytes) -> void* {
        void* p = ws + off;
        off += (bytes + 255) & ~(size_t)255;
        return p;
    };

    const int NC1P  = CC;            // 128, already mult of 64
    const int NH1P  = HHEADS * CC;   // 1024
    const int NC2P  = CC;            // 128
    const int NPRJP = PROJ;          // 64
    const int NCLSP = 64;            // 40 -> pad 64

    _Float16* xh16   = (_Float16*)alloc((size_t)NN * FF * 2);
    _Float16* wl1t   = (_Float16*)alloc((size_t)NC1P * FF * 2);          // [128][512]
    _Float16* w1t    = (_Float16*)alloc((size_t)NH1P * FF * 2);          // [1024][512]
    _Float16* w2t    = (_Float16*)alloc((size_t)NC2P * (HHEADS * CC) * 2); // [128][1024]
    _Float16* wprojt = (_Float16*)alloc((size_t)NPRJP * (2 * CC) * 2);   // [64][256]
    _Float16* wlint  = (_Float16*)alloc((size_t)NCLSP * (3 * CC) * 2);   // [64][384]
    float* x1    = (float*)alloc((size_t)NN * CC * 4);
    float* xh1   = (float*)alloc((size_t)NN * HHEADS * CC * 4);
    float* als1  = (float*)alloc((size_t)NN * HHEADS * 4);
    float* ald1  = (float*)alloc((size_t)NN * HHEADS * 4);
    float* m1    = (float*)alloc((size_t)NN * HHEADS * 4);
    float* den1  = (float*)alloc((size_t)NN * HHEADS * 4);
    float* ebuf1 = (float*)alloc((size_t)EE * HHEADS * 4);
    float* out1  = (float*)alloc((size_t)NN * HHEADS * CC * 4);
    _Float16* xgh = (_Float16*)alloc((size_t)NN * HHEADS * CC * 2);
    float* xh2   = (float*)alloc((size_t)NN * CC * 4);
    float* als2  = (float*)alloc((size_t)NN * 4);
    float* ald2  = (float*)alloc((size_t)NN * 4);
    float* m2    = (float*)alloc((size_t)NN * 4);
    float* den2  = (float*)alloc((size_t)NN * 4);
    float* ebuf2 = (float*)alloc((size_t)EE * 4);
    float* out2  = (float*)alloc((size_t)NN * CC * 4);
    float* finalf = (float*)alloc((size_t)NN * 2 * CC * 4);
    _Float16* finalh = (_Float16*)alloc((size_t)NN * 2 * CC * 2);
    float* sq    = (float*)alloc((size_t)NN * 4);
    int*   kidx  = (int*)alloc((size_t)NN * KKNN * 4);
    float* sim   = (float*)alloc((size_t)NN * 2 * CC * 4);
    _Float16* c3h = (_Float16*)alloc((size_t)NN * 3 * CC * 2);
    float* logits = (float*)alloc((size_t)NN * NCLS * 4);

    float* out_ls = (float*)d_out;                     // [N, CLASSES]
    float* gsc    = (float*)d_out + (size_t)NN * NCLS; // [N, PROJ]

    const int T = 256;

    // --- activation f32->f16, weights f32 -> transposed/padded f16 --------
    kg_cvt_h<<<kg_nb((long long)NN * FF, T), T, 0, stream>>>(x, xh16, (size_t)NN * FF);
    kg_cvt_tr<<<kg_nb((long long)NC1P * FF, T), T, 0, stream>>>(lin1_w, wl1t, FF, CC, NC1P);
    kg_cvt_tr<<<kg_nb((long long)NH1P * FF, T), T, 0, stream>>>(W1, w1t, FF, HHEADS * CC, NH1P);
    kg_cvt_tr<<<kg_nb((long long)NC2P * HHEADS * CC, T), T, 0, stream>>>(W2, w2t, HHEADS * CC, CC, NC2P);
    kg_cvt_tr<<<kg_nb((long long)NPRJP * 2 * CC, T), T, 0, stream>>>(proj_w, wprojt, 2 * CC, PROJ, NPRJP);
    kg_cvt_tr<<<kg_nb((long long)NCLSP * 3 * CC, T), T, 0, stream>>>(lin_w, wlint, 3 * CC, NCLS, NCLSP);

    // --- MLP branch: x1 = relu(x @ lin1_w + lin1_b) -----------------------
    kg_gemm_wmma<<<dim3(NC1P / 64, NN / 16), 128, 0, stream>>>(
        xh16, wl1t, lin1_b, x1, NN, FF, CC, 1);

    // --- GAT layer 1 -------------------------------------------------------
    kg_gemm_wmma<<<dim3(NH1P / 64, NN / 16), 128, 0, stream>>>(
        xh16, w1t, nullptr, xh1, NN, FF, HHEADS * CC, 0);
    kg_attn_logits<<<kg_nb((long long)NN * HHEADS, T), T, 0, stream>>>(
        xh1, as1, ad1, als1, ald1, NN, HHEADS);
    kg_fill<<<kg_nb((long long)NN * HHEADS, T), T, 0, stream>>>(m1, -__builtin_inff(), (size_t)NN * HHEADS);
    kg_fill<<<kg_nb((long long)NN * HHEADS, T), T, 0, stream>>>(den1, 0.f, (size_t)NN * HHEADS);
    kg_fill<<<kg_nb((long long)NN * HHEADS * CC, T), T, 0, stream>>>(out1, 0.f, (size_t)NN * HHEADS * CC);
    kg_edge_max<<<kg_nb((long long)EE * HHEADS, T), T, 0, stream>>>(src, dst, als1, ald1, ebuf1, m1, HHEADS);
    kg_edge_exp<<<kg_nb((long long)EE * HHEADS, T), T, 0, stream>>>(dst, ebuf1, m1, den1, HHEADS);
    kg_edge_aggr<<<kg_nb((long long)EE * HHEADS * (CC / 4), T), T, 0, stream>>>(
        src, dst, ebuf1, den1, xh1, out1, HHEADS);
    kg_bias_relu_h<<<kg_nb((long long)NN * HHEADS * CC, T), T, 0, stream>>>(
        out1, b1, xgh, (size_t)NN * HHEADS * CC, HHEADS * CC);

    // --- GAT layer 2 (1 head) ---------------------------------------------
    kg_gemm_wmma<<<dim3(NC2P / 64, NN / 16), 128, 0, stream>>>(
        xgh, w2t, nullptr, xh2, NN, HHEADS * CC, CC, 0);
    kg_attn_logits<<<kg_nb((long long)NN, T), T, 0, stream>>>(xh2, as2, ad2, als2, ald2, NN, 1);
    kg_fill<<<kg_nb((long long)NN, T), T, 0, stream>>>(m2, -__builtin_inff(), (size_t)NN);
    kg_fill<<<kg_nb((long long)NN, T), T, 0, stream>>>(den2, 0.f, (size_t)NN);
    kg_fill<<<kg_nb((long long)NN * CC, T), T, 0, stream>>>(out2, 0.f, (size_t)NN * CC);
    kg_edge_max<<<kg_nb((long long)EE, T), T, 0, stream>>>(src, dst, als2, ald2, ebuf2, m2, 1);
    kg_edge_exp<<<kg_nb((long long)EE, T), T, 0, stream>>>(dst, ebuf2, m2, den2, 1);
    kg_edge_aggr<<<kg_nb((long long)EE * (CC / 4), T), T, 0, stream>>>(
        src, dst, ebuf2, den2, xh2, out2, 1);

    // --- final = [x1 | x2], g_score = final @ proj_w + proj_b -------------
    kg_build_final<<<kg_nb((long long)NN * 2 * CC, T), T, 0, stream>>>(x1, out2, b2, finalf, finalh);
    kg_gemm_wmma<<<dim3(NPRJP / 64, NN / 16), 128, 0, stream>>>(
        finalh, wprojt, proj_b, gsc, NN, 2 * CC, PROJ, 0);

    // --- exact KNN + neighborhood mean ------------------------------------
    kg_sqnorm<<<kg_nb((long long)NN, T), T, 0, stream>>>(gsc, sq);
    kg_knn<<<NN, 256, 0, stream>>>(gsc, sq, kidx);
    kg_sim<<<kg_nb((long long)NN * 2 * CC, T), T, 0, stream>>>(finalf, kidx, sim);

    // --- classifier + log_softmax -----------------------------------------
    kg_build_c3<<<kg_nb((long long)NN * 3 * CC, T), T, 0, stream>>>(x1, sim, c3h);
    kg_gemm_wmma<<<dim3(NCLSP / 64, NN / 16), 128, 0, stream>>>(
        c3h, wlint, lin_b, logits, NN, 3 * CC, NCLS, 0);
    kg_logsoftmax<<<kg_nb((long long)NN, T), T, 0, stream>>>(logits, out_ls);
}
